// LlamaDecoderLayer_71313636983298
// MI455X (gfx1250) — matrix operand order repaired and checked by reading, not compiled, not run on hardware
//
#include <hip/hip_runtime.h>
#include <hip/hip_bf16.h>
#include <math.h>

#define Bb 4
#define Ll 1024
#define Hh 2048
#define NHh 16
#define NKVv 4
#define DHh 128
#define Ee 8
#define TOPKk 2
#define Dd 2048
#define NT (Bb*Ll)
#define EPSf 1e-6f
#define MINVf (-1e15f)

typedef __attribute__((ext_vector_type(16))) __bf16 bf16x16;
typedef __attribute__((ext_vector_type(8)))  __bf16 bf16x8;
typedef __attribute__((ext_vector_type(4)))  __bf16 bf16x4;
typedef __attribute__((ext_vector_type(8)))  float  f32x8;
typedef __attribute__((ext_vector_type(4)))  float  f32x4;

// ---------------- RMSNorm ----------------
__global__ __launch_bounds__(256) void rmsnorm_kernel(const float* __restrict__ x,
                                                      const float* __restrict__ w,
                                                      float* __restrict__ y, int ncol) {
  int row = blockIdx.x;
  const float* xr = x + (long)row * ncol;
  float* yr = y + (long)row * ncol;
  __shared__ float red[256];
  float s = 0.f;
  for (int c = threadIdx.x; c < ncol; c += 256) { float v = xr[c]; s += v * v; }
  red[threadIdx.x] = s; __syncthreads();
  for (int o = 128; o > 0; o >>= 1) {
    if (threadIdx.x < o) red[threadIdx.x] += red[threadIdx.x + o];
    __syncthreads();
  }
  float inv = rsqrtf(red[0] / (float)ncol + EPSf);
  for (int c = threadIdx.x; c < ncol; c += 256) yr[c] = w[c] * xr[c] * inv;
}

// ---------------- Generic batched WMMA GEMM ----------------
// C[M,N] = A[M,K]*B[K,N], fp32 in/out, v_wmma_f32_16x16x32_bf16 compute.
// TRANSB: B memory is [N,K] row-major (used for Q*K^T).
// MODE 0: plain strided batch; MODE 1: MoE segment + row gather; MODE 2: MoE segment, packed A.
struct GemmP {
  const float* A; const float* Bm; float* C;
  int M, N, K;
  int lda, ldb, ldc;
  int zdivA, hdivA; long sA1, sA2;
  int zdivB, hdivB; long sB1, sB2;
  int zdivC, hdivC; long sC1, sC2;
  const int* segOff;
  const int* counts;
  const int* rowmap;
};

#define BM 64
#define BN 128
#define BK 32
#define AST 40   // LDS row pitch (elements): 80B, 16B-aligned rows
#define BSTK 40

template <bool TRANSB, int MODE>
__global__ __launch_bounds__(256) void gemm_bf16_kernel(GemmP p) {
  __shared__ __bf16 As[BM * AST];    // [m][k]
  __shared__ __bf16 Bs[BN * BSTK];   // [n][k]  (transposed staging)
  int z = blockIdx.z;
  int tid = threadIdx.x;

  long boff = (long)(z / p.zdivB) * p.sB1 + (long)((z % p.zdivB) / p.hdivB) * p.sB2;
  long aoff = 0, coff = 0;
  int Mz = p.M;
  const int* rmap = nullptr;
  if (MODE != 0) {
    int so = p.segOff[z];
    Mz = p.counts[z];
    coff = (long)so * p.ldc;
    if (MODE == 1) rmap = p.rowmap + so;
    else aoff = (long)so * p.lda;
  } else {
    aoff = (long)(z / p.zdivA) * p.sA1 + (long)((z % p.zdivA) / p.hdivA) * p.sA2;
    coff = (long)(z / p.zdivC) * p.sC1 + (long)((z % p.zdivC) / p.hdivC) * p.sC2;
  }
  int m0 = blockIdx.y * BM;
  if (m0 >= Mz) return;
  int n0 = blockIdx.x * BN;

  int wid = tid >> 5, lane = tid & 31;
  int half = lane >> 4, ln = lane & 15;
  int wm = wid & 1, wn = wid >> 1;

  f32x8 zero8 = {0.f, 0.f, 0.f, 0.f, 0.f, 0.f, 0.f, 0.f};
  f32x8 acc[2][2];
  acc[0][0] = zero8; acc[0][1] = zero8; acc[1][0] = zero8; acc[1][1] = zero8;

  const f32x4 vzero = {0.f, 0.f, 0.f, 0.f};

  for (int k0 = 0; k0 < p.K; k0 += BK) {
    // ---- Stage A tile (64x32): float4 loads -> bf16x4 LDS stores, branch-free guard
#pragma unroll
    for (int it = 0; it < 2; it++) {
      int i = tid + it * 256;          // 0..511
      int r = i >> 3;                  // 0..63
      int c4 = (i & 7) * 4;            // 0..28
      int gr = m0 + r;
      int grc = gr < Mz ? gr : Mz - 1; // clamp; Mz>=1 here
      long arow = (MODE == 1) ? (long)rmap[grc] : (long)grc;
      f32x4 v = *(const f32x4*)(p.A + aoff + arow * p.lda + (k0 + c4));
      if (gr >= Mz) v = vzero;
      *(bf16x4*)(&As[r * AST + c4]) = __builtin_convertvector(v, bf16x4);
    }
    // ---- Stage B tile (32x128) into transposed LDS [n][k]
    if (TRANSB) {
      // B memory [N,K]: contiguous k per n -> fully vectorized both sides
#pragma unroll
      for (int it = 0; it < 4; it++) {
        int i = tid + it * 256;        // 0..1023
        int c = i >> 3;                // n 0..127
        int r4 = (i & 7) * 4;          // k 0..28
        f32x4 v = *(const f32x4*)(p.Bm + boff + (long)(n0 + c) * p.ldb + (k0 + r4));
        *(bf16x4*)(&Bs[c * BSTK + r4]) = __builtin_convertvector(v, bf16x4);
      }
    } else {
      // B memory [K,N]: coalesced float4 along n, transpose into LDS
#pragma unroll
      for (int it = 0; it < 4; it++) {
        int i = tid + it * 256;
        int r = i >> 5;                // k 0..31
        int c4 = (i & 31) * 4;         // n 0..124
        f32x4 v = *(const f32x4*)(p.Bm + boff + (long)(k0 + r) * p.ldb + (n0 + c4));
        Bs[(c4 + 0) * BSTK + r] = (__bf16)v.x;
        Bs[(c4 + 1) * BSTK + r] = (__bf16)v.y;
        Bs[(c4 + 2) * BSTK + r] = (__bf16)v.z;
        Bs[(c4 + 3) * BSTK + r] = (__bf16)v.w;
      }
    }
    // Prefetch next K-tile (global_prefetch_b8)
    if (k0 + BK < p.K) {
      __builtin_prefetch(p.A + aoff + (long)(m0 + (tid >> 3)) * p.lda + (k0 + BK), 0, 1);
      if (TRANSB)
        __builtin_prefetch(p.Bm + boff + (long)(n0 + (tid >> 1)) * p.ldb + (k0 + BK), 0, 1);
      else
        __builtin_prefetch(p.Bm + boff + (long)(k0 + BK + (tid >> 3)) * p.ldb + n0, 0, 1);
    }
    __syncthreads();

    // ---- Fragments: contiguous 16B/32B LDS loads, no element shuffling
    bf16x16 af[2], bfm[2];
#pragma unroll
    for (int t = 0; t < 2; t++) {
      const __bf16* arow = &As[(wm * 32 + t * 16 + ln) * AST];
      bf16x8 alo = *(const bf16x8*)(arow + 8 * half);        // k = 8h .. 8h+7
      bf16x8 ahi = *(const bf16x8*)(arow + 16 + 8 * half);   // k = 16+8h .. 16+8h+7
      af[t] = __builtin_shufflevector(alo, ahi, 0, 1, 2, 3, 4, 5, 6, 7,
                                      8, 9, 10, 11, 12, 13, 14, 15);
      const __bf16* brow = &Bs[(wn * 32 + t * 16 + ln) * BSTK];
      bf16x8 blo = *(const bf16x8*)(brow + 8 * half);        // k = 8h .. 8h+7
      bf16x8 bhi = *(const bf16x8*)(brow + 16 + 8 * half);   // k = 16+8h .. 16+8h+7
      bfm[t] = __builtin_shufflevector(blo, bhi, 0, 1, 2, 3, 4, 5, 6, 7,
                                       8, 9, 10, 11, 12, 13, 14, 15);
    }
#pragma unroll
    for (int i = 0; i < 2; i++)
#pragma unroll
      for (int j = 0; j < 2; j++)
        acc[i][j] = __builtin_amdgcn_wmma_f32_16x16x32_bf16(
            false, af[i], false, bfm[j], (short)0, acc[i][j], false, false);
    __syncthreads();
  }

#pragma unroll
  for (int i = 0; i < 2; i++) {
    int rb = m0 + wm * 32 + i * 16 + half * 8;
#pragma unroll
    for (int j = 0; j < 2; j++) {
      int cb = n0 + wn * 32 + j * 16 + ln;
#pragma unroll
      for (int r = 0; r < 8; r++) {
        int rr = rb + r;
        if (rr < Mz) p.C[coff + (long)rr * p.ldc + cb] = acc[i][j][r];
      }
    }
  }
}

// ---------------- RoPE (in place, tables on the fly) ----------------
__global__ void rope_kernel(float* __restrict__ qk, int nheads) {
  long i = (long)blockIdx.x * 256 + threadIdx.x;
  long total = (long)NT * nheads * 64;
  if (i >= total) return;
  int d = (int)(i % 64);
  long tmp = i / 64;
  int hh = (int)(tmp % nheads);
  int t = (int)(tmp / nheads);
  int pos = t % Ll;
  float inv = powf(10000.f, -(2.f * d) / 128.f);
  float fr = (float)pos * inv;
  float cs = cosf(fr), sn = sinf(fr);
  float* pp = qk + (long)t * nheads * DHh + (long)hh * DHh;
  float x0 = pp[d], x1 = pp[d + 64];
  pp[d] = x0 * cs - x1 * sn;
  pp[d + 64] = x1 * cs + x0 * sn;
}

// ---------------- Causal masked softmax (in place) ----------------
__global__ __launch_bounds__(256) void softmax_kernel(float* __restrict__ scores,
                                                      const int* __restrict__ amask) {
  int row = blockIdx.x;                // B*NH*L rows of length L
  int qpos = row % Ll;
  int b = row / (NHh * Ll);
  float* sr = scores + (long)row * Ll;
  const int* mrow = amask + (long)b * Ll;
  const float scale = 0.08838834764831845f;  // 1/sqrt(128)
  __shared__ float red[256];
  float mx = -3.4e38f;
  for (int c = threadIdx.x; c < Ll; c += 256) {
    float v = sr[c] * scale;
    if (c > qpos) v += MINVf;
    if (mrow[c] == 0) v += MINVf;
    sr[c] = v;
    mx = fmaxf(mx, v);
  }
  red[threadIdx.x] = mx; __syncthreads();
  for (int o = 128; o > 0; o >>= 1) {
    if (threadIdx.x < o) red[threadIdx.x] = fmaxf(red[threadIdx.x], red[threadIdx.x + o]);
    __syncthreads();
  }
  mx = red[0]; __syncthreads();
  float sm = 0.f;
  for (int c = threadIdx.x; c < Ll; c += 256) { float e = expf(sr[c] - mx); sr[c] = e; sm += e; }
  red[threadIdx.x] = sm; __syncthreads();
  for (int o = 128; o > 0; o >>= 1) {
    if (threadIdx.x < o) red[threadIdx.x] += red[threadIdx.x + o];
    __syncthreads();
  }
  float inv = 1.f / red[0];
  for (int c = threadIdx.x; c < Ll; c += 256) sr[c] *= inv;
}

// ---------------- Elementwise ----------------
__global__ void add2_kernel(float* __restrict__ y, const float* __restrict__ a, long n) {
  long i = (long)blockIdx.x * 256 + threadIdx.x;
  if (i < n) y[i] += a[i];
}
__global__ void add3_kernel(float* __restrict__ o, const float* __restrict__ a,
                            const float* __restrict__ b, const float* __restrict__ c, long n) {
  long i = (long)blockIdx.x * 256 + threadIdx.x;
  if (i < n) o[i] = a[i] + b[i] + c[i];
}
__global__ void silu_mul_kernel(float* __restrict__ o, const float* __restrict__ g,
                                const float* __restrict__ u, long n) {
  long i = (long)blockIdx.x * 256 + threadIdx.x;
  if (i < n) { float gv = g[i]; o[i] = (gv / (1.f + expf(-gv))) * u[i]; }
}

// ---------------- Gating ----------------
__global__ __launch_bounds__(256) void gate_logits_kernel(const float* __restrict__ x2,
                                                          const float* __restrict__ gw,
                                                          const float* __restrict__ gb,
                                                          float* __restrict__ logits) {
  int t = blockIdx.x;
  int wid = threadIdx.x >> 5, lane = threadIdx.x & 31;  // wave per expert (wave32)
  const float* xr = x2 + (long)t * Hh;
  float s = 0.f;
  for (int c = lane; c < Hh; c += 32) s += xr[c] * gw[(long)c * Ee + wid];
  for (int o = 16; o > 0; o >>= 1) s += __shfl_down(s, o, 32);
  if (lane == 0) logits[(long)t * Ee + wid] = s + gb[wid];
}

__global__ void topk_kernel(const float* __restrict__ logits, int* __restrict__ eid,
                            float* __restrict__ gval) {
  int t = blockIdx.x * 256 + threadIdx.x;
  if (t >= NT) return;
  const float* lr = logits + (long)t * Ee;
  float l[8];
#pragma unroll
  for (int e = 0; e < 8; e++) l[e] = lr[e];
  int i0 = 0;
  for (int e = 1; e < 8; e++) if (l[e] > l[i0]) i0 = e;
  int i1 = -1;
  for (int e = 0; e < 8; e++) { if (e == i0) continue; if (i1 < 0 || l[e] > l[i1]) i1 = e; }
  float m = fmaxf(l[i0], l[i1]);
  float e0 = expf(l[i0] - m), e1 = expf(l[i1] - m);
  float inv = 1.f / (e0 + e1);
  eid[t * 2] = i0; eid[t * 2 + 1] = i1;
  gval[t * 2] = e0 * inv; gval[t * 2 + 1] = e1 * inv;
}

// ---------------- Routing (top-2 token gather) ----------------
__global__ void route_init_kernel(int* counts, int* cursor) {
  if (threadIdx.x < Ee) { counts[threadIdx.x] = 0; cursor[threadIdx.x] = 0; }
}
__global__ void route_count_kernel(const int* __restrict__ eid, int* counts) {
  int i = blockIdx.x * 256 + threadIdx.x;
  if (i < NT * TOPKk) atomicAdd(&counts[eid[i]], 1);
}
__global__ void route_scan_kernel(const int* __restrict__ counts, int* offs) {
  if (threadIdx.x == 0) { int s = 0; for (int e = 0; e < Ee; e++) { offs[e] = s; s += counts[e]; } }
}
__global__ void route_assign_kernel(const int* __restrict__ eid, const int* __restrict__ offs,
                                    int* cursor, int* rowmap, int* tokrow) {
  int i = blockIdx.x * 256 + threadIdx.x;
  if (i >= NT * TOPKk) return;
  int e = eid[i];
  int pos = atomicAdd(&cursor[e], 1);
  int idx = offs[e] + pos;
  rowmap[idx] = i >> 1;     // token id
  tokrow[i] = idx;          // inverse map for deterministic combine
}

__global__ void combine_kernel(const float* __restrict__ y, const int* __restrict__ tokrow,
                               const float* __restrict__ gval, float* __restrict__ moe) {
  long i = (long)blockIdx.x * 256 + threadIdx.x;
  if (i >= (long)NT * Hh) return;
  int t = (int)(i / Hh), c = (int)(i % Hh);
  int r0 = tokrow[t * 2], r1 = tokrow[t * 2 + 1];
  moe[i] = gval[t * 2] * y[(long)r0 * Hh + c] + gval[t * 2 + 1] * y[(long)r1 * Hh + c];
}

// ---------------- Launch ----------------
extern "C" void kernel_launch(void* const* d_in, const int* in_sizes, int n_in,
                              void* d_out, int out_size, void* d_ws, size_t ws_size,
                              hipStream_t stream) {
  (void)in_sizes; (void)n_in; (void)out_size; (void)ws_size;
  const float* hidden = (const float*)d_in[0];
  const int*   amask  = (const int*)d_in[1];
  const float* ln1w = (const float*)d_in[2];
  const float* wq = (const float*)d_in[3];
  const float* wk = (const float*)d_in[4];
  const float* wv = (const float*)d_in[5];
  const float* wo = (const float*)d_in[6];
  const float* ln2w = (const float*)d_in[7];
  const float* gatew = (const float*)d_in[8];
  const float* gateb = (const float*)d_in[9];
  const float* Wg = (const float*)d_in[10];
  const float* Wu = (const float*)d_in[11];
  const float* Wd = (const float*)d_in[12];
  const float* sg = (const float*)d_in[13];
  const float* su = (const float*)d_in[14];
  const float* sd = (const float*)d_in[15];
  float* out = (float*)d_out;
  float* logits = out + (size_t)NT * Hh;

  float* ws = (float*)d_ws;
  size_t off = 0;
  auto alloc = [&](size_t n) { float* p = ws + off; off += n; return p; };
  float* xn1 = alloc((size_t)NT * Hh);            // reused as moe output later
  float* q   = alloc((size_t)NT * Hh);
  float* kb  = alloc((size_t)NT * NKVv * DHh);
  float* vb  = alloc((size_t)NT * NKVv * DHh);
  float* o_  = alloc((size_t)NT * Hh);
  float* hb  = alloc((size_t)NT * Hh);
  float* xn2 = alloc((size_t)NT * Hh);
  float* scores = alloc((size_t)Bb * NHh * Ll * Ll);  // 67.1M floats, reused for MoE bufs
  float* gbuf = scores;
  float* ubuf = gbuf + (size_t)NT * TOPKk * Dd;
  float* abuf = ubuf + (size_t)NT * TOPKk * Dd;
  float* ybuf = abuf + (size_t)NT * TOPKk * Dd;       // also reused as shared-expert out
  float* gval = alloc((size_t)NT * TOPKk);
  int* eid    = (int*)alloc((size_t)NT * TOPKk);
  int* rowmap = (int*)alloc((size_t)NT * TOPKk);
  int* tokrow = (int*)alloc((size_t)NT * TOPKk);
  int* counts = (int*)alloc(64);
  int* cursor = counts + 8;
  int* offs   = counts + 16;
  float* moe  = xn1;

  // transB: 0/1; mode: 0 plain, 1 seg+gather, 2 seg+packed
  auto gemm = [&](const float* A, const float* Bm, float* C, int M, int N, int K,
                  int lda, int ldb, int ldc, int batch,
                  int zdA, int hdA, long sA1, long sA2,
                  int zdB, int hdB, long sB1, long sB2,
                  int zdC, int hdC, long sC1, long sC2,
                  int transB, int mode,
                  const int* segO, const int* cnts, const int* rmap, int maxM) {
    GemmP p;
    p.A = A; p.Bm = Bm; p.C = C; p.M = M; p.N = N; p.K = K;
    p.lda = lda; p.ldb = ldb; p.ldc = ldc;
    p.zdivA = zdA; p.hdivA = hdA; p.sA1 = sA1; p.sA2 = sA2;
    p.zdivB = zdB; p.hdivB = hdB; p.sB1 = sB1; p.sB2 = sB2;
    p.zdivC = zdC; p.hdivC = hdC; p.sC1 = sC1; p.sC2 = sC2;
    p.segOff = segO; p.counts = cnts; p.rowmap = rmap;
    dim3 g(N / BN, (maxM + BM - 1) / BM, batch);
    if (transB)          gemm_bf16_kernel<true, 0><<<g, 256, 0, stream>>>(p);
    else if (mode == 0)  gemm_bf16_kernel<false, 0><<<g, 256, 0, stream>>>(p);
    else if (mode == 1)  gemm_bf16_kernel<false, 1><<<g, 256, 0, stream>>>(p);
    else                 gemm_bf16_kernel<false, 2><<<g, 256, 0, stream>>>(p);
  };

  long nTok = (long)NT * Hh;

  // ---- Attention ----
  rmsnorm_kernel<<<NT, 256, 0, stream>>>(hidden, ln1w, xn1, Hh);
  gemm(xn1, wq, q,  NT, NHh * DHh, Hh, Hh, NHh * DHh, NHh * DHh, 1,
       1,1,0,0, 1,1,0,0, 1,1,0,0, 0, 0, nullptr, nullptr, nullptr, NT);
  gemm(xn1, wk, kb, NT, NKVv * DHh, Hh, Hh, NKVv * DHh, NKVv * DHh, 1,
       1,1,0,0, 1,1,0,0, 1,1,0,0, 0, 0, nullptr, nullptr, nullptr, NT);
  gemm(xn1, wv, vb, NT, NKVv * DHh, Hh, Hh, NKVv * DHh, NKVv * DHh, 1,
       1,1,0,0, 1,1,0,0, 1,1,0,0, 0, 0, nullptr, nullptr, nullptr, NT);

  rope_kernel<<<(NT * NHh * 64 + 255) / 256, 256, 0, stream>>>(q, NHh);
  rope_kernel<<<(NT * NKVv * 64 + 255) / 256, 256, 0, stream>>>(kb, NKVv);

  // scores[b,h] = q[b,:,h,:] @ k[b,:,h/4,:]^T   (GQA via hdivB=4)
  gemm(q, kb, scores, Ll, Ll, DHh, NHh * DHh, NKVv * DHh, Ll, Bb * NHh,
       NHh, 1, (long)Ll * NHh * DHh, DHh,
       NHh, 4, (long)Ll * NKVv * DHh, DHh,
       1, 1, (long)Ll * Ll, 0,
       1, 0, nullptr, nullptr, nullptr, Ll);

  softmax_kernel<<<Bb * NHh * Ll, 256, 0, stream>>>(scores, amask);

  // o[b,:,h,:] = probs[b,h] @ v[b,:,h/4,:]
  gemm(scores, vb, o_, Ll, DHh, Ll, Ll, NKVv * DHh, NHh * DHh, Bb * NHh,
       1, 1, (long)Ll * Ll, 0,
       NHh, 4, (long)Ll * NKVv * DHh, DHh,
       NHh, 1, (long)Ll * NHh * DHh, DHh,
       0, 0, nullptr, nullptr, nullptr, Ll);

  gemm(o_, wo, hb, NT, Hh, Hh, Hh, Hh, Hh, 1,
       1,1,0,0, 1,1,0,0, 1,1,0,0, 0, 0, nullptr, nullptr, nullptr, NT);
  add2_kernel<<<(int)((nTok + 255) / 256), 256, 0, stream>>>(hb, hidden, nTok);

  // ---- MoE ----
  rmsnorm_kernel<<<NT, 256, 0, stream>>>(hb, ln2w, xn2, Hh);
  gate_logits_kernel<<<NT, 256, 0, stream>>>(xn2, gatew, gateb, logits);
  topk_kernel<<<NT / 256, 256, 0, stream>>>(logits, eid, gval);

  route_init_kernel<<<1, 256, 0, stream>>>(counts, cursor);
  route_count_kernel<<<(NT * TOPKk) / 256, 256, 0, stream>>>(eid, counts);
  route_scan_kernel<<<1, 32, 0, stream>>>(counts, offs);
  route_assign_kernel<<<(NT * TOPKk) / 256, 256, 0, stream>>>(eid, offs, cursor, rowmap, tokrow);

  // Expert GEMMs: static grid, per-expert M from counts[], gathered A rows.
  gemm(xn2, Wg, gbuf, NT, Dd, Hh, Hh, Dd, Dd, Ee,
       1,1,0,0, 1,1,(long)Hh * Dd,0, 1,1,0,0, 0, 1, offs, counts, rowmap, NT);
  gemm(xn2, Wu, ubuf, NT, Dd, Hh, Hh, Dd, Dd, Ee,
       1,1,0,0, 1,1,(long)Hh * Dd,0, 1,1,0,0, 0, 1, offs, counts, rowmap, NT);
  silu_mul_kernel<<<(int)(((long)NT * TOPKk * Dd + 255) / 256), 256, 0, stream>>>(
      abuf, gbuf, ubuf, (long)NT * TOPKk * Dd);
  gemm(abuf, Wd, ybuf, NT, Hh, Dd, Dd, Hh, Hh, Ee,
       1,1,0,0, 1,1,(long)Dd * Hh,0, 1,1,0,0, 0, 2, offs, counts, nullptr, NT);
  combine_kernel<<<(int)((nTok + 255) / 256), 256, 0, stream>>>(ybuf, tokrow, gval, moe);

  // ---- Shared expert (reuses gbuf/ubuf/abuf/ybuf) ----
  gemm(xn2, sg, gbuf, NT, Dd, Hh, Hh, Dd, Dd, 1,
       1,1,0,0, 1,1,0,0, 1,1,0,0, 0, 0, nullptr, nullptr, nullptr, NT);
  gemm(xn2, su, ubuf, NT, Dd, Hh, Hh, Dd, Dd, 1,
       1,1,0,0, 1,1,0,0, 1,1,0,0, 0, 0, nullptr, nullptr, nullptr, NT);
  silu_mul_kernel<<<(int)((nTok + 255) / 256), 256, 0, stream>>>(abuf, gbuf, ubuf, nTok);
  gemm(abuf, sd, ybuf, NT, Hh, Dd, Dd, Hh, Hh, 1,
       1,1,0,0, 1,1,0,0, 1,1,0,0, 0, 0, nullptr, nullptr, nullptr, NT);

  // out = moe + shared + residual
  add3_kernel<<<(int)((nTok + 255) / 256), 256, 0, stream>>>(out, moe, ybuf, hb, nTok);
}
